// LangModel_56573309223169
// MI455X (gfx1250) — matrix operand order repaired and checked by reading, not compiled, not run on hardware
//
#include <hip/hip_runtime.h>
#include <hip/hip_bf16.h>
#include <math.h>
#include <stdint.h>

// ---------------------------------------------------------------------------
// Types for CDNA5 WMMA (wave32):  v_wmma_f32_16x16x32_bf16
//   A: 16x32 bf16 (16 bf16 per lane), B: 32x16 bf16, C/D: 16x16 f32 (8 VGPRs)
// ---------------------------------------------------------------------------
typedef __bf16 bf16_t;
typedef __attribute__((ext_vector_type(16))) __bf16 v16bf;
typedef __attribute__((ext_vector_type(8)))  float  v8f;

__device__ __forceinline__ bf16_t to_bf16(float f) { return (bf16_t)f; }
__device__ __forceinline__ int imin(int a, int b) { return a < b ? a : b; }

// CDNA5 async global->LDS copy (ASYNCcnt-tracked DMA, bypasses VGPRs).
__device__ __forceinline__ void async_ld_b128(uint32_t lds_byte_off, const void* gaddr) {
  asm volatile("global_load_async_to_lds_b128 %0, %1, off"
               :: "v"(lds_byte_off), "v"(gaddr) : "memory");
}
__device__ __forceinline__ void wait_async0() {
  asm volatile("s_wait_asynccnt 0x0" ::: "memory");
}

// ---------------------------------------------------------------------------
// Fragment loaders.
// A (16x32 bf16) layout used here (contiguous k, not the ISA table order): lane l -> row M = l%16,
//   elem i = K kb+i, kb = (l/16)*16.
// B (32x16 bf16) layout used here (same contiguous k order as A): lane l -> col N = l%16,
//   elem i = K kb+i, kb = (l/16)*16.
// ---------------------------------------------------------------------------

template <bool VEC4>
__device__ __forceinline__ v16bf load_A_fast(const float* __restrict__ p) {
  if (VEC4) p = (const float*)__builtin_assume_aligned(p, 16);
  v16bf a;
#pragma unroll
  for (int i = 0; i < 8; ++i) {
    a[i]     = to_bf16(p[i]);
    a[i + 8] = to_bf16(p[i + 8]);
  }
  return a;
}

// Guarded A: clamp index in-bounds, select 0 afterwards (v_cndmask, no branch).
__device__ __forceinline__ v16bf load_A_guard(const float* __restrict__ A, long lda,
                                              int m0, int k0, int kmax, int lane) {
  const int row = m0 + (lane & 15);
  const int kb  = k0 + ((lane >> 4) << 4);
  const float* p = A + (long)row * lda;
  v16bf a;
#pragma unroll
  for (int i = 0; i < 8; ++i) {
    const int ka = kb + i, kc = kb + 8 + i;
    const float va = p[imin(ka, kmax - 1)];
    const float vc = p[imin(kc, kmax - 1)];
    a[i]     = (ka < kmax) ? to_bf16(va) : (bf16_t)0.0f;
    a[i + 8] = (kc < kmax) ? to_bf16(vc) : (bf16_t)0.0f;
  }
  return a;
}

// Guarded B, K-major [K x ldb]: clamp k and n, select 0.
__device__ __forceinline__ v16bf load_B_kmajor_guard(const float* __restrict__ B, long ldb,
                                                     int k0, int n0, int kmax, int nmax,
                                                     int lane) {
  const int n  = n0 + (lane & 15);
  const int nc = imin(n, nmax - 1);
  const int kb = k0 + ((lane >> 4) << 4);
  const bool nok = (n < nmax);
  v16bf b;
#pragma unroll
  for (int i = 0; i < 16; ++i) {
    const int k = kb + i;
    const float v = B[(long)imin(k, kmax - 1) * ldb + nc];
    b[i] = (nok && k < kmax) ? to_bf16(v) : (bf16_t)0.0f;
  }
  return b;
}

// Guarded B, transposed storage W = B^T [N x ldb]: clamp n row and k, select 0.
__device__ __forceinline__ v16bf load_B_trans_guard(const float* __restrict__ W, long ldb,
                                                    int k0, int n0, int kmax, int nmax,
                                                    int lane) {
  const int n  = n0 + (lane & 15);
  const int kb = k0 + ((lane >> 4) << 4);
  const bool nok = (n < nmax);
  const float* p = W + (long)imin(n, nmax - 1) * ldb;
  v16bf b;
#pragma unroll
  for (int i = 0; i < 16; ++i) {
    const int k = kb + i;
    const float v = p[imin(k, kmax - 1)];
    b[i] = (nok && k < kmax) ? to_bf16(v) : (bf16_t)0.0f;
  }
  return b;
}

// B fragment from an LDS tile stored [64 n-rows x 36 floats] (BTRANS kernels).
// 36*i mod 64 distinct for i=0..15 => conflict-free ds_load_b128 per lane.
#define BT_STRIDE 36
__device__ __forceinline__ v16bf load_B_lds_trans(const float* __restrict__ tile,
                                                  int j, int lane) {
  const int n  = j * 16 + (lane & 15);   // block-local column 0..63
  const int kb = (lane >> 4) << 4;       // 0 or 16 within the 32-K chunk
  const float* p =
      (const float*)__builtin_assume_aligned(tile + n * BT_STRIDE + kb, 16);
  v16bf b;
#pragma unroll
  for (int i = 0; i < 16; ++i) b[i] = to_bf16(p[i]);
  return b;
}

// B fragment from an LDS tile stored [32 k-rows x 68 floats] (K-major kernel).
// Consecutive lanes read consecutive addresses => conflict-free ds_load_b32.
#define KT_STRIDE 68
__device__ __forceinline__ v16bf load_B_lds_kmajor(const float* __restrict__ tile,
                                                   int j, int lane) {
  const int n  = j * 16 + (lane & 15);   // block-local column 0..63
  const int kb = (lane >> 4) << 4;       // 0 or 16 within the 32-K chunk
  const float* p = tile + kb * KT_STRIDE + n;
  v16bf b;
#pragma unroll
  for (int i = 0; i < 16; ++i) b[i] = to_bf16(p[i * KT_STRIDE]);
  return b;
}

// ---------------------------------------------------------------------------
// GEMM for B in K-major storage [K x ldb] with 16B-aligned rows (K1).
// Interior blocks DMA the 32x64 B tile into LDS (row segments are contiguous
// and aligned), shared by all 4 waves. A stays on scalar global loads (odd lda).
// ---------------------------------------------------------------------------
__global__ __launch_bounds__(128) void gemm_bf16_wmma_kmajor(
    const float* __restrict__ A, long lda, int KAmax,
    const float* __restrict__ B, long ldb, int KBmax,
    float* __restrict__ C, long ldc,
    int N, int K, int Nstore)
{
  __shared__ float btile[32 * KT_STRIDE];  // 8704 B, padded k-rows

  const int lane = threadIdx.x & 31;
  const int wave = threadIdx.x >> 5;
  const int m0   = blockIdx.y * 64 + wave * 16;
  const int nblk = blockIdx.x * 64;

  v8f acc[4] = {};

  const bool n_edge = (nblk + 64 > N);
  const int  kfast  = n_edge ? 0 : (imin(imin(KAmax, KBmax), K) & ~31);

  const int arow = m0 + (lane & 15);
  const float* __restrict__ aptr = A + (long)arow * lda + ((lane >> 4) << 4);

  // Staging map: thread tid -> k-row tid&31, 16-float quarter (tid>>5)*16.
  const int skrow = threadIdx.x & 31;
  const int sq    = (threadIdx.x >> 5) << 4;
  const uint32_t ldsoff =
      (uint32_t)(uintptr_t)(&btile[skrow * KT_STRIDE + sq]);
  const float* __restrict__ bsrc = B + (long)skrow * ldb + nblk + sq;

  for (int k0 = 0; k0 < kfast; k0 += 32) {
    // Async-DMA this K-chunk's 32x64 B tile (4x b128 per thread, coalesced).
    const float* g = bsrc + (long)k0 * ldb;
    async_ld_b128(ldsoff +  0, g + 0);
    async_ld_b128(ldsoff + 16, g + 4);
    async_ld_b128(ldsoff + 32, g + 8);
    async_ld_b128(ldsoff + 48, g + 12);

    __builtin_prefetch(aptr + k0 + 64, 0, 1);       // global_prefetch_b8
    const v16bf a = load_A_fast<false>(aptr + k0);  // overlaps the DMA

    wait_async0();        // s_wait_asynccnt 0
    __syncthreads();      // tile visible to all 4 waves

#pragma unroll
    for (int j = 0; j < 4; ++j) {
      const v16bf b = load_B_lds_kmajor(btile, j, lane);
      acc[j] = __builtin_amdgcn_wmma_f32_16x16x32_bf16(
          false, a, false, b, (short)0, acc[j], false, false);
    }
    __syncthreads();      // don't overwrite the tile until everyone is done
  }

  // Edge path: K tail and/or last-column block (branch-free clamped loads).
  for (int k0 = kfast; k0 < K; k0 += 32) {
    const v16bf a = load_A_guard(A, lda, m0, k0, KAmax, lane);
#pragma unroll
    for (int j = 0; j < 4; ++j) {
      const v16bf b = load_B_kmajor_guard(B, ldb, k0, nblk + j * 16, KBmax, N, lane);
      acc[j] = __builtin_amdgcn_wmma_f32_16x16x32_bf16(
          false, a, false, b, (short)0, acc[j], false, false);
    }
  }

#pragma unroll
  for (int j = 0; j < 4; ++j) {
    const int n = nblk + j * 16 + (lane & 15);
    if (n >= Nstore) continue;
    const int mb = m0 + ((lane >> 4) << 3);
#pragma unroll
    for (int r = 0; r < 8; ++r) C[(long)(mb + r) * ldc + n] = acc[j][r];
  }
}

// ---------------------------------------------------------------------------
// GEMM for B given as B^T [N x ldb], rows 16B-aligned (K2 and K4).
// Interior blocks stage the shared 32x64 B tile in LDS with CDNA5 async
// global->LDS DMA (ASYNCcnt), giving 4x reuse across the block's waves and
// overlapping the DMA with the A-fragment global loads.
// ---------------------------------------------------------------------------
__global__ __launch_bounds__(128) void gemm_bf16_wmma_btrans(
    const float* __restrict__ A, long lda, int KAmax,
    const float* __restrict__ B, long ldb, int KBmax,
    const float* __restrict__ bias1, const float* __restrict__ bias2,
    float* __restrict__ C, long ldc,
    int N, int K, int Nstore)
{
  __shared__ float btile[64 * BT_STRIDE];  // 9216 B, padded n-rows

  const int lane = threadIdx.x & 31;
  const int wave = threadIdx.x >> 5;
  const int m0   = blockIdx.y * 64 + wave * 16;
  const int nblk = blockIdx.x * 64;

  v8f acc[4] = {};

  const bool n_edge = (nblk + 64 > N);
  const int  kfast  = n_edge ? 0 : (imin(imin(KAmax, KBmax), K) & ~31);

  const int arow = m0 + (lane & 15);
  const float* __restrict__ aptr = A + (long)arow * lda + ((lane >> 4) << 4);

  // Staging map: thread tid -> n-row tid&63, 16-float half (tid>>6)*16.
  const int srow  = threadIdx.x & 63;
  const int shalf = (threadIdx.x >> 6) << 4;
  const uint32_t ldsoff =
      (uint32_t)(uintptr_t)(&btile[srow * BT_STRIDE + shalf]);
  const float* __restrict__ bsrc = B + (long)(nblk + srow) * ldb + shalf;

  for (int k0 = 0; k0 < kfast; k0 += 32) {
    // Async-DMA this K-chunk's 32x64 B tile (4x b128 per thread).
    const float* g = bsrc + k0;
    async_ld_b128(ldsoff +  0, g + 0);
    async_ld_b128(ldsoff + 16, g + 4);
    async_ld_b128(ldsoff + 32, g + 8);
    async_ld_b128(ldsoff + 48, g + 12);

    __builtin_prefetch(aptr + k0 + 64, 0, 1);     // global_prefetch_b8
    const v16bf a = load_A_fast<true>(aptr + k0); // overlaps the DMA

    wait_async0();        // s_wait_asynccnt 0
    __syncthreads();      // tile visible to all 4 waves

#pragma unroll
    for (int j = 0; j < 4; ++j) {
      const v16bf b = load_B_lds_trans(btile, j, lane);
      acc[j] = __builtin_amdgcn_wmma_f32_16x16x32_bf16(
          false, a, false, b, (short)0, acc[j], false, false);
    }
    __syncthreads();      // don't overwrite the tile until everyone is done
  }

  // Edge path: K tail and/or last-column block (branch-free clamped loads).
  for (int k0 = kfast; k0 < K; k0 += 32) {
    const v16bf a = load_A_guard(A, lda, m0, k0, KAmax, lane);
#pragma unroll
    for (int j = 0; j < 4; ++j) {
      const v16bf b = load_B_trans_guard(B, ldb, k0, nblk + j * 16, KBmax, N, lane);
      acc[j] = __builtin_amdgcn_wmma_f32_16x16x32_bf16(
          false, a, false, b, (short)0, acc[j], false, false);
    }
  }

  // Store: 16x16 f32 C layout -> lane l holds N = l%16, M = r + 8*(l/16).
#pragma unroll
  for (int j = 0; j < 4; ++j) {
    const int n = nblk + j * 16 + (lane & 15);
    if (n >= Nstore) continue;
    float bv = 0.0f;
    if (n < N) {
      if (bias1) bv += bias1[n];
      if (bias2) bv += bias2[n];
    }
    const int mb = m0 + ((lane >> 4) << 3);
#pragma unroll
    for (int r = 0; r < 8; ++r) {
      C[(long)(mb + r) * ldc + n] = acc[j][r] + bv;
    }
  }
}

// ---------------------------------------------------------------------------
// LSTM recurrence: strictly sequential over S=2048 steps -> single workgroup,
// 1024 threads (32 wave32s on one WGP), h in LDS, c in registers.
// Thread t computes gate rows t and t+1024:
//   t in [0,512)    -> i-gate[t] and g-gate[t]
//   t in [512,1024) -> f-gate[t-512] and o-gate[t-512]
// f/o pre-activations exchanged through LDS each step.
// ---------------------------------------------------------------------------
#define S_LEN 2048
#define HDIM  512
#define G4    2048   // 4*H

__global__ __launch_bounds__(1024) void lstm_seq_kernel(
    const float* __restrict__ xw,    // [S, 4H] = x_emb@w_ih^T + b_ih + b_hh
    const float* __restrict__ w_hh,  // [4H, H] row-major (rows 16B aligned)
    const float* __restrict__ mask,  // [S, H] inverted-dropout mask
    float* __restrict__ hs)          // [S, H] h after dropout (fed to logits)
{
  __shared__ float h_sh[HDIM];
  __shared__ float fo_sh[2 * HDIM];

  const int t = threadIdx.x;
  for (int j = t; j < HDIM; j += 1024) h_sh[j] = 0.0f;
  float c = 0.0f;  // valid for t < HDIM
  __syncthreads();

  const int g0 = t;          // i (t<512) or f (t>=512)
  const int g1 = t + 1024;   // g (t<512) or o (t>=512)
  const float* __restrict__ w0 =
      (const float*)__builtin_assume_aligned(w_hh + (long)g0 * HDIM, 16);
  const float* __restrict__ w1 =
      (const float*)__builtin_assume_aligned(w_hh + (long)g1 * HDIM, 16);

  for (int s = 0; s < S_LEN; ++s) {
    float acc0 = xw[(long)s * G4 + g0];
    float acc1 = xw[(long)s * G4 + g1];
#pragma unroll 8
    for (int k = 0; k < HDIM; ++k) {
      const float hk = h_sh[k];
      acc0 = fmaf(w0[k], hk, acc0);
      acc1 = fmaf(w1[k], hk, acc1);
    }
    if (t >= HDIM) {
      fo_sh[t - HDIM]        = acc0;  // f pre-activation
      fo_sh[t - HDIM + HDIM] = acc1;  // o pre-activation
    }
    __syncthreads();
    if (t < HDIM) {
      const float ig = 1.0f / (1.0f + __expf(-acc0));
      const float gg = tanhf(acc1);
      const float fg = 1.0f / (1.0f + __expf(-fo_sh[t]));
      const float og = 1.0f / (1.0f + __expf(-fo_sh[t + HDIM]));
      c = fg * c + ig * gg;
      float h = og * tanhf(c);
      h *= mask[(long)s * HDIM + t];
      h_sh[t] = h;
      hs[(long)s * HDIM + t] = h;
    }
    __syncthreads();
  }
}

// ---------------------------------------------------------------------------
// Problem constants
// ---------------------------------------------------------------------------
#define SEQ   2048
#define VOCAB 50257
#define EMB   300
#define EPAD  320     // EMB padded to a multiple of 32 for the K2 K-loop

extern "C" void kernel_launch(void* const* d_in, const int* in_sizes, int n_in,
                              void* d_out, int out_size, void* d_ws, size_t ws_size,
                              hipStream_t stream) {
  const float* inputs = (const float*)d_in[0];   // [S, V]
  const float* emb    = (const float*)d_in[1];   // [V, E]
  const float* w_ih   = (const float*)d_in[2];   // [4H, E]
  const float* w_hh   = (const float*)d_in[3];   // [4H, H]
  const float* b_ih   = (const float*)d_in[4];   // [4H]
  const float* b_hh   = (const float*)d_in[5];   // [4H]
  const float* w_lin  = (const float*)d_in[6];   // [V, H]
  const float* b_lin  = (const float*)d_in[7];   // [V]
  const float* mask   = (const float*)d_in[8];   // [S, H]
  float* out = (float*)d_out;                    // [S, V]

  // Workspace layout (f32):
  float* x_emb = (float*)d_ws;                    // [S, EPAD]  (pad cols zeroed)
  float* xw    = x_emb + (size_t)SEQ * EPAD;      // [S, 4H]
  float* hs    = xw    + (size_t)SEQ * G4;        // [S, H]

  const dim3 blk(128);

  // K1: x_emb[S, EPAD] = inputs[S,V] @ emb[V,E]   (async-LDS B tiles)
  {
    dim3 grid(EPAD / 64, SEQ / 64);
    gemm_bf16_wmma_kmajor<<<grid, blk, 0, stream>>>(
        inputs, (long)VOCAB, VOCAB,
        emb,    (long)EMB,   VOCAB,
        x_emb, (long)EPAD,
        /*N=*/EMB, /*K=*/VOCAB, /*Nstore=*/EPAD);
  }

  // K2: xw[S, 4H] = x_emb[S, EPAD] @ w_ih^T + b_ih + b_hh  (async-LDS B tiles)
  {
    dim3 grid(G4 / 64, SEQ / 64);
    gemm_bf16_wmma_btrans<<<grid, blk, 0, stream>>>(
        x_emb, (long)EPAD, EPAD,
        w_ih,  (long)EMB,  EMB,
        b_ih, b_hh,
        xw, (long)G4,
        /*N=*/G4, /*K=*/EPAD, /*Nstore=*/G4);
  }

  // K3: sequential LSTM recurrence with dropout on h feeding back
  lstm_seq_kernel<<<1, 1024, 0, stream>>>(xw, w_hh, mask, hs);

  // K4: out[S, V] = hs[S, H] @ w_lin^T + b_lin  (async-LDS B tiles)
  {
    dim3 grid((VOCAB + 63) / 64, SEQ / 64);
    gemm_bf16_wmma_btrans<<<grid, blk, 0, stream>>>(
        hs,    (long)HDIM, HDIM,
        w_lin, (long)HDIM, HDIM,
        b_lin, nullptr,
        out, (long)VOCAB,
        /*N=*/VOCAB, /*K=*/HDIM, /*Nstore=*/VOCAB);
  }
}
